// MambaBlock_53652731462297
// MI455X (gfx1250) — matrix operand order repaired and checked by reading, not compiled, not run on hardware
//
#include <hip/hip_runtime.h>
#include <hip/hip_bf16.h>

// ---------------------------------------------------------------------------
// Mamba block for MI455X (gfx1250, wave32, WMMA).
// LN -> GEMM(W_in)+SiLU split -> GEMM(W_del) -> GEMM(W_dt)+softplus
//    -> GEMM(W_B), GEMM(W_C) -> selective scan (fused gate epilogue)
//    -> GEMM(W_out)+bias+residual.
// GEMMs: v_wmma_f32_16x16x32_bf16, double-buffered LDS staging via
// GLOBAL_LOAD_ASYNC_TO_LDS_B128 (ASYNCcnt) when the builtins exist.
// ---------------------------------------------------------------------------

#define HID 1024
#define SD  16
#define EXPD 2048
#define DTR 64
#define B_  4
#define T_  2048
#define M_  (B_ * T_)   // 8192 tokens

typedef __bf16 bf16;
typedef __attribute__((ext_vector_type(16))) __bf16 v16bf;
typedef __attribute__((ext_vector_type(8)))  __bf16 v8bf;
typedef __attribute__((ext_vector_type(8)))  float  v8f;
typedef int i32x4 __attribute__((__vector_size__(16)));

#define AS1 __attribute__((address_space(1)))
#define AS3 __attribute__((address_space(3)))

#if defined(__gfx1250__) && \
    __has_builtin(__builtin_amdgcn_global_load_async_to_lds_b128) && \
    __has_builtin(__builtin_amdgcn_s_wait_asynccnt)
#define HAS_ASYNC_LDS 1
#else
#define HAS_ASYNC_LDS 0
#endif

// ---------------------------------------------------------------------------
// Weight prep: W (K x N, f32, row-major) -> Wt (N x K, bf16, row-major)
// ---------------------------------------------------------------------------
__global__ __launch_bounds__(256)
void wtrans_kernel(const float* __restrict__ W, bf16* __restrict__ Wt,
                   int K, int N) {
  __shared__ float tile[32][33];
  int kb = blockIdx.y * 32, nb = blockIdx.x * 32;
  for (int i = threadIdx.y; i < 32; i += 8) {
    int k = kb + i, n = nb + threadIdx.x;
    tile[i][threadIdx.x] = (k < K && n < N) ? W[(size_t)k * N + n] : 0.f;
  }
  __syncthreads();
  for (int i = threadIdx.y; i < 32; i += 8) {
    int n = nb + i, k = kb + threadIdx.x;
    if (n < N && k < K) Wt[(size_t)n * K + k] = (bf16)tile[threadIdx.x][i];
  }
}

// ---------------------------------------------------------------------------
// LayerNorm over HID, one block (256 threads = 8 waves) per token row.
// ---------------------------------------------------------------------------
__global__ __launch_bounds__(256)
void ln_kernel(const float* __restrict__ x, const float* __restrict__ g,
               const float* __restrict__ b, bf16* __restrict__ xn) {
  const int row = blockIdx.x;
  const float* xr = x + (size_t)row * HID;
  float v[HID / 256];
  float s = 0.f, s2 = 0.f;
#pragma unroll
  for (int i = 0; i < HID / 256; i++) {
    v[i] = xr[threadIdx.x + i * 256];
    s += v[i]; s2 += v[i] * v[i];
  }
#pragma unroll
  for (int off = 16; off; off >>= 1) {
    s  += __shfl_xor(s,  off, 32);
    s2 += __shfl_xor(s2, off, 32);
  }
  __shared__ float rs[8], rs2[8];
  int wid = threadIdx.x >> 5, lane = threadIdx.x & 31;
  if (lane == 0) { rs[wid] = s; rs2[wid] = s2; }
  __syncthreads();
  s = 0.f; s2 = 0.f;
#pragma unroll
  for (int i = 0; i < 8; i++) { s += rs[i]; s2 += rs2[i]; }
  const float mean = s * (1.f / HID);
  const float var  = s2 * (1.f / HID) - mean * mean;
  const float inv  = rsqrtf(var + 1e-5f);
#pragma unroll
  for (int i = 0; i < HID / 256; i++) {
    int c = threadIdx.x + i * 256;
    xn[(size_t)row * HID + c] = (bf16)((v[i] - mean) * inv * g[c] + b[c]);
  }
}

// ---------------------------------------------------------------------------
// Epilogue
// ---------------------------------------------------------------------------
struct Epi {
  const float* bias;   // length N (or 2*EXPD for split mode)
  const float* res;    // residual, MxN f32 (mode 4)
  float* o32;
  bf16*  o16;
  bf16*  o16b;         // gate half (mode 1)
  int    mode;         // 0 f32, 1 silu-split bf16, 2 softplus f32, 3 bf16, 4 f32+res
  int    split;        // EXPD for mode 1
};

__device__ __forceinline__ void epi_store(float vv, int r, int c,
                                          const Epi& ep, int N) {
  float v = vv + (ep.bias ? ep.bias[c] : 0.f);
  switch (ep.mode) {
    case 0:
      ep.o32[(size_t)r * N + c] = v;
      break;
    case 1: {
      float s = v / (1.f + __expf(-v));   // silu
      if (c < ep.split) ep.o16 [(size_t)r * ep.split + c]              = (bf16)s;
      else              ep.o16b[(size_t)r * ep.split + (c - ep.split)] = (bf16)s;
      break;
    }
    case 2: {
      float sp = (v > 20.f) ? v : log1pf(__expf(v));  // softplus
      ep.o32[(size_t)r * N + c] = sp;
      break;
    }
    case 3:
      ep.o16[(size_t)r * N + c] = (bf16)v;
      break;
    case 4:
      ep.o32[(size_t)r * N + c] = v + ep.res[(size_t)r * N + c];
      break;
  }
}

// ---------------------------------------------------------------------------
// Tiled WMMA GEMM, compile-time K, double-buffered async LDS staging.
// C = epilogue(A[MxKK] @ Bt[NxKK]^T + bias); A/Bt bf16 row-major, ld = KK.
// ---------------------------------------------------------------------------
template <int BM, int BN, int BK, int WROWS, int WCOLS, int WM, int WN, int KK>
__global__ __launch_bounds__(WROWS * WCOLS * 32)
void gemm_bf16_kernel(const bf16* __restrict__ A, const bf16* __restrict__ Bt,
                      int M, int N, Epi ep) {
  static_assert(BM == WROWS * WM * 16, "BM mismatch");
  static_assert(BN == WCOLS * WN * 16, "BN mismatch");
  static_assert(KK % BK == 0, "K tiling");
  constexpr int NT     = WROWS * WCOLS * 32;
  constexpr int BKP    = BK + 8;           // 80B row stride: 16B aligned, no 4-way conflicts
  constexpr int KITERS = KK / BK;
  constexpr int CA     = BM * (BK / 8);    // 16B chunks in A tile
  constexpr int CB     = BN * (BK / 8);
  __shared__ bf16 lA[2][BM][BKP];
  __shared__ bf16 lB[2][BN][BKP];

  const int tid  = threadIdx.x;
  const int wave = tid >> 5;
  const int lane = tid & 31;
  const int wr   = wave % WROWS;
  const int wc   = wave / WROWS;
  const int bm   = blockIdx.y * BM;
  const int bn   = blockIdx.x * BN;
  const int rrow = lane & 15;
  const int hi   = (lane >= 16) ? 1 : 0;

  v8f acc[WM][WN] = {};

  // Per-tile stage: compile-time unrolled, 16B per lane per chunk.
  auto stage = [&](int buf, int k0) {
#pragma unroll
    for (int j = 0; j < (CA + NT - 1) / NT; j++) {
      int idx = tid + j * NT;
      if ((CA % NT == 0) || idx < CA) {
        int r = idx / (BK / 8), c = (idx % (BK / 8)) * 8;
        const bf16* src = A + (size_t)(bm + r) * KK + k0 + c;
#if HAS_ASYNC_LDS
        __builtin_amdgcn_global_load_async_to_lds_b128(
            (AS1 i32x4*)src, (AS3 i32x4*)&lA[buf][r][c], 0, 0);
#else
        *(v8bf*)&lA[buf][r][c] = *(const v8bf*)src;
#endif
      }
    }
#pragma unroll
    for (int j = 0; j < (CB + NT - 1) / NT; j++) {
      int idx = tid + j * NT;
      if ((CB % NT == 0) || idx < CB) {
        int r = idx / (BK / 8), c = (idx % (BK / 8)) * 8;
        const bf16* src = Bt + (size_t)(bn + r) * KK + k0 + c;
#if HAS_ASYNC_LDS
        __builtin_amdgcn_global_load_async_to_lds_b128(
            (AS1 i32x4*)src, (AS3 i32x4*)&lB[buf][r][c], 0, 0);
#else
        *(v8bf*)&lB[buf][r][c] = *(const v8bf*)src;
#endif
      }
    }
  };

  stage(0, 0);
  int buf = 0;
#pragma unroll 1
  for (int it = 0; it < KITERS; it++) {
#if HAS_ASYNC_LDS
    __builtin_amdgcn_s_wait_asynccnt(0);   // my tile writes done
#endif
    __syncthreads();                        // everyone's tile writes visible
    if (it + 1 < KITERS) stage(buf ^ 1, (it + 1) * BK);

    // ---- fragments (ISA 16-bit A 16x32 / B 32x16 wave32 layouts) ----
    v16bf afr[WM], bfr[WN];
#pragma unroll
    for (int m = 0; m < WM; m++) {
      int row = wr * WM * 16 + m * 16 + rrow;
      int ka  = hi ? 8 : 0;                 // lanes>=16: K 8..15 / 24..31
      v8bf lo = *(const v8bf*)&lA[buf][row][ka];
      v8bf hh = *(const v8bf*)&lA[buf][row][ka + 16];
      afr[m]  = __builtin_shufflevector(lo, hh,
                  0,1,2,3,4,5,6,7,8,9,10,11,12,13,14,15);
    }
#pragma unroll
    for (int n = 0; n < WN; n++) {
      int col = wc * WN * 16 + n * 16 + rrow;
      int kb  = hi ? 8 : 0;                 // lanes>=16: K 8..15 / 24..31
      v8bf lo = *(const v8bf*)&lB[buf][col][kb];
      v8bf hh = *(const v8bf*)&lB[buf][col][kb + 16];
      bfr[n]  = __builtin_shufflevector(lo, hh,
                  0,1,2,3,4,5,6,7,8,9,10,11,12,13,14,15);
    }
#pragma unroll
    for (int m = 0; m < WM; m++)
#pragma unroll
      for (int n = 0; n < WN; n++)
        acc[m][n] = __builtin_amdgcn_wmma_f32_16x16x32_bf16(
            false, afr[m], false, bfr[n], (short)0, acc[m][n], false, false);

    __syncthreads();                        // tile reads done before overwrite
    buf ^= 1;
  }

  // ---- store: D layout = lane col L%16, rows i + 8*(L>=16) ----
#pragma unroll
  for (int m = 0; m < WM; m++)
#pragma unroll
    for (int n = 0; n < WN; n++) {
      int cbase = bn + wc * WN * 16 + n * 16 + rrow;
      int rbase = bm + wr * WM * 16 + m * 16 + hi * 8;
#pragma unroll
      for (int i = 0; i < 8; i++)
        epi_store(acc[m][n][i], rbase + i, cbase, ep, N);
    }
}

// ---------------------------------------------------------------------------
// Selective scan: one thread per (batch, channel); 16-state recurrence in
// registers; fused  y = (scan + u*Dp) * silu(gate)  epilogue -> bf16.
// ---------------------------------------------------------------------------
__global__ __launch_bounds__(256)
void scan_kernel(const bf16* __restrict__ u, const float* __restrict__ delta,
                 const float* __restrict__ Bm, const float* __restrict__ Cm,
                 const float* __restrict__ A_log, const float* __restrict__ Dp,
                 const bf16* __restrict__ sg, bf16* __restrict__ y) {
  const int e = blockIdx.x * 256 + threadIdx.x;   // 0..EXPD-1
  const int b = blockIdx.y;                       // 0..B_-1
  float A[SD], h[SD];
#pragma unroll
  for (int s = 0; s < SD; s++) {
    A[s] = -__expf(A_log[(size_t)e * SD + s]);
    h[s] = 0.f;
  }
  const float dp = Dp[e];
  for (int t = 0; t < T_; t++) {
    const size_t bt = (size_t)b * T_ + t;
    const float  d  = delta[bt * EXPD + e];
    const float  uu = (float)u[bt * EXPD + e];
    float Bv[SD], Cv[SD];
#pragma unroll
    for (int q = 0; q < SD / 4; q++) {            // wave-uniform broadcasts
      float4 bq = *(const float4*)(Bm + bt * SD + 4 * q);
      float4 cq = *(const float4*)(Cm + bt * SD + 4 * q);
      Bv[4*q] = bq.x; Bv[4*q+1] = bq.y; Bv[4*q+2] = bq.z; Bv[4*q+3] = bq.w;
      Cv[4*q] = cq.x; Cv[4*q+1] = cq.y; Cv[4*q+2] = cq.z; Cv[4*q+3] = cq.w;
    }
    const float du = d * uu;
    float yt = 0.f;
#pragma unroll
    for (int s = 0; s < SD; s++) {
      h[s] = __expf(d * A[s]) * h[s] + du * Bv[s];
      yt  += Cv[s] * h[s];
    }
    const float out = (yt + uu * dp) * (float)sg[bt * EXPD + e];
    y[bt * EXPD + e] = (bf16)out;
  }
}

// ---------------------------------------------------------------------------
// Host-side launch
// ---------------------------------------------------------------------------
extern "C" void kernel_launch(void* const* d_in, const int* in_sizes, int n_in,
                              void* d_out, int out_size, void* d_ws, size_t ws_size,
                              hipStream_t stream) {
  (void)in_sizes; (void)n_in; (void)out_size; (void)ws_size;
  const float* x     = (const float*)d_in[0];
  const float* ln_g  = (const float*)d_in[1];
  const float* ln_b  = (const float*)d_in[2];
  const float* W_in  = (const float*)d_in[3];
  const float* b_in  = (const float*)d_in[4];
  const float* W_del = (const float*)d_in[5];
  const float* b_del = (const float*)d_in[6];
  const float* W_dt  = (const float*)d_in[7];
  const float* b_dt  = (const float*)d_in[8];
  const float* W_B   = (const float*)d_in[9];
  const float* b_B   = (const float*)d_in[10];
  const float* W_C   = (const float*)d_in[11];
  const float* b_C   = (const float*)d_in[12];
  const float* A_log = (const float*)d_in[13];
  const float* Dp    = (const float*)d_in[14];
  const float* W_out = (const float*)d_in[15];
  const float* b_out = (const float*)d_in[16];
  float* out = (float*)d_out;

  char* ws = (char*)d_ws;
  size_t off = 0;
  auto alloc = [&](size_t bytes) -> void* {
    void* p = ws + off;
    off = (off + bytes + 255) & ~(size_t)255;
    return p;
  };
  bf16*  xnb   = (bf16*) alloc((size_t)M_ * HID * 2);
  bf16*  u_b   = (bf16*) alloc((size_t)M_ * EXPD * 2);
  bf16*  sg_b  = (bf16*) alloc((size_t)M_ * EXPD * 2);
  bf16*  d1_b  = (bf16*) alloc((size_t)M_ * DTR * 2);
  float* deltp = (float*)alloc((size_t)M_ * EXPD * 4);
  float* Bmp   = (float*)alloc((size_t)M_ * SD * 4);
  float* Cmp   = (float*)alloc((size_t)M_ * SD * 4);
  bf16*  y_b   = (bf16*) alloc((size_t)M_ * EXPD * 2);
  bf16*  WinT  = (bf16*) alloc((size_t)HID * 2 * EXPD * 2);
  bf16*  WdelT = (bf16*) alloc((size_t)EXPD * DTR * 2);
  bf16*  WdtT  = (bf16*) alloc((size_t)DTR * EXPD * 2);
  bf16*  WBT   = (bf16*) alloc((size_t)EXPD * SD * 2);
  bf16*  WCT   = (bf16*) alloc((size_t)EXPD * SD * 2);
  bf16*  WoutT = (bf16*) alloc((size_t)EXPD * HID * 2);

  dim3 tb(32, 8);
  auto tg = [](int K, int N) { return dim3((N + 31) / 32, (K + 31) / 32); };
  wtrans_kernel<<<tg(HID,  2*EXPD), tb, 0, stream>>>(W_in,  WinT,  HID,  2*EXPD);
  wtrans_kernel<<<tg(EXPD, DTR),    tb, 0, stream>>>(W_del, WdelT, EXPD, DTR);
  wtrans_kernel<<<tg(DTR,  EXPD),   tb, 0, stream>>>(W_dt,  WdtT,  DTR,  EXPD);
  wtrans_kernel<<<tg(EXPD, SD),     tb, 0, stream>>>(W_B,   WBT,   EXPD, SD);
  wtrans_kernel<<<tg(EXPD, SD),     tb, 0, stream>>>(W_C,   WCT,   EXPD, SD);
  wtrans_kernel<<<tg(EXPD, HID),    tb, 0, stream>>>(W_out, WoutT, EXPD, HID);

  ln_kernel<<<M_, 256, 0, stream>>>(x, ln_g, ln_b, xnb);

  // xp = xn @ W_in + b_in ; u = silu(lo), sg = silu(hi)   (M x 4096, K=1024)
  {
    Epi ep{b_in, nullptr, nullptr, u_b, sg_b, 1, EXPD};
    gemm_bf16_kernel<128, 128, 32, 4, 2, 2, 4, HID>
        <<<dim3((2 * EXPD) / 128, M_ / 128), 256, 0, stream>>>(
            xnb, WinT, M_, 2 * EXPD, ep);
  }
  // d1 = u @ W_del + b_del  (M x 64, K=2048) -> bf16
  {
    Epi ep{b_del, nullptr, nullptr, d1_b, nullptr, 3, 0};
    gemm_bf16_kernel<128, 64, 32, 4, 2, 2, 2, EXPD>
        <<<dim3(DTR / 64, M_ / 128), 256, 0, stream>>>(u_b, WdelT, M_, DTR, ep);
  }
  // delta = softplus(d1 @ W_dt + b_dt)  (M x 2048, K=64) -> f32
  {
    Epi ep{b_dt, nullptr, deltp, nullptr, nullptr, 2, 0};
    gemm_bf16_kernel<128, 128, 32, 4, 2, 2, 4, DTR>
        <<<dim3(EXPD / 128, M_ / 128), 256, 0, stream>>>(d1_b, WdtT, M_, EXPD, ep);
  }
  // Bm = u @ W_B + b_B ; Cm = u @ W_C + b_C  (M x 16, K=2048) -> f32
  {
    Epi ep{b_B, nullptr, Bmp, nullptr, nullptr, 0, 0};
    gemm_bf16_kernel<128, 16, 32, 8, 1, 1, 1, EXPD>
        <<<dim3(1, M_ / 128), 256, 0, stream>>>(u_b, WBT, M_, SD, ep);
  }
  {
    Epi ep{b_C, nullptr, Cmp, nullptr, nullptr, 0, 0};
    gemm_bf16_kernel<128, 16, 32, 8, 1, 1, 1, EXPD>
        <<<dim3(1, M_ / 128), 256, 0, stream>>>(u_b, WCT, M_, SD, ep);
  }
  // selective scan + fused (y + u*Dp) * silu(gate) -> y_b (bf16)
  scan_kernel<<<dim3(EXPD / 256, B_), 256, 0, stream>>>(
      u_b, deltp, Bmp, Cmp, A_log, Dp, sg_b, y_b);

  // out = y @ W_out + b_out + x  (M x 1024, K=2048) -> f32
  {
    Epi ep{b_out, x, out, nullptr, nullptr, 4, 0};
    gemm_bf16_kernel<128, 128, 32, 4, 2, 2, 4, EXPD>
        <<<dim3(HID / 128, M_ / 128), 256, 0, stream>>>(y_b, WoutT, M_, HID, ep);
  }
}
